// ScaledMultiHeadAttention_52725018526318
// MI455X (gfx1250) — matrix operand order repaired and checked by reading, not compiled, not run on hardware
//
#include <hip/hip_runtime.h>
#include <hip/hip_bf16.h>

#define DEV __device__ __forceinline__

typedef __attribute__((ext_vector_type(16))) __bf16 v16bf;
typedef __attribute__((ext_vector_type(8)))  float  v8f;
typedef __attribute__((ext_vector_type(4)))  unsigned int u32x4;
typedef __attribute__((ext_vector_type(8)))  int          i32x8;
typedef __attribute__((ext_vector_type(4)))  int          i32x4;

union FragB16 { v16bf v; unsigned int u[8]; };

// ---- constants for this problem ----
#define BB 2
#define TT 2048
#define CC 2048
#define HH 16
#define HD 128
#define MM (BB*TT)      // 4096
#define N_QKV (3*CC)    // 6144

// Compiler-level memory clobber (effective only for escaped memory).
#define LDS_CLOBBER() asm volatile("" ::: "memory")
// Escape an LDS pointer so alias analysis must assume the TDM writes it.
#define LDS_ESCAPE(p) asm volatile("" :: "v"(p) : "memory")

DEV unsigned short f2bf(float f) {
  union { float f; unsigned int u; } x; x.f = f;
  unsigned int r = x.u + 0x7FFFu + ((x.u >> 16) & 1u);
  return (unsigned short)(r >> 16);
}
DEV float bf2f(unsigned short h) {
  union { unsigned int u; float f; } x; x.u = ((unsigned int)h) << 16;
  return x.f;
}

DEV v8f wmma_bf16(const FragB16& a, const FragB16& b, v8f c) {
  return __builtin_amdgcn_wmma_f32_16x16x32_bf16(
      /*neg_a=*/false, a.v, /*neg_b=*/false, b.v,
      /*c_mod=*/(short)0, c, /*reuse_a=*/false, /*reuse_b=*/false);
}

// -------------------- TDM: async 2-D tile load global -> LDS --------------------
// Loads a tile of 16 dwords (32 bf16) x 128 rows, row stride = stride_dw dwords,
// into LDS at lds_byte_off. LDS padding: +4 dwords after every 64 bytes
// (16 dwords) -> LDS row stride of 20 dwords = 40 bf16 (bank-conflict-free).
DEV void tdm_load_tile(unsigned lds_byte_off, const unsigned short* gptr,
                       unsigned stride_dw) {
  unsigned long long ga = (unsigned long long)gptr;
  u32x4 g0;
  g0.x = 1u;                                               // count = 1 descriptor
  g0.y = lds_byte_off;                                     // lds_addr
  g0.z = (unsigned)ga;                                     // global_addr[31:0]
  g0.w = (unsigned)((ga >> 32) & 0x01FFFFFFu) | (2u << 30);// global_addr[56:32], type=2
  i32x8 g1;
  // data_size=2 (4B), pad_enable=1, pad_interval=3 (64B), pad_amount=3 (4 dwords)
  g1[0] = (int)((2u << 16) | (1u << 20) | (3u << 22) | (3u << 25));
  g1[1] = (int)((stride_dw & 0xFFFFu) << 16);              // tensor_dim0[15:0]
  g1[2] = (int)((stride_dw >> 16) & 0xFFFFu);              // tensor_dim0[31:16], dim1 lo = 0
  g1[3] = (int)(0x10u | (16u << 16));                      // tensor_dim1 = 1M, tile_dim0 = 16 dw
  g1[4] = 128;                                             // tile_dim1 = 128 rows, tile_dim2 = 0
  g1[5] = (int)stride_dw;                                  // tensor_dim0_stride[31:0]
  g1[6] = 0;                                               // stride hi / dim1_stride lo
  g1[7] = 0;
  i32x4 z4 = {0, 0, 0, 0};
  i32x8 z8 = {0, 0, 0, 0, 0, 0, 0, 0};
  __builtin_amdgcn_tensor_load_to_lds(g0, g1, z4, z4, z8, 0);
  LDS_CLOBBER();
}

// -------------------- fp32 -> bf16 convert --------------------
__global__ void k_f32_to_bf16(const float* __restrict__ in,
                              unsigned short* __restrict__ out, int n) {
  int i = blockIdx.x * blockDim.x + threadIdx.x;
  int stride = gridDim.x * blockDim.x;
  for (; i < n; i += stride) out[i] = f2bf(in[i]);
}

// -------------------- WMMA GEMM: C[M,N] = A[M,K] * W[N,K]^T --------------------
// A, W: bf16 row-major. Output: fp32 (outF) or bf16 (outB).
// Block tile 128(M) x 128(N), K step 32. 8 waves, each 32x64 (2x4 wmma tiles).
// Tiles staged by the Tensor Data Mover, double-buffered (TENSORcnt).
#define ABUF (128 * 40)   // elems per (padded) tile buffer
__global__ __launch_bounds__(256)
void k_gemm_bf16(const unsigned short* __restrict__ A,
                 const unsigned short* __restrict__ W,
                 float* __restrict__ outF, unsigned short* __restrict__ outB,
                 int M, int N, int K) {
  __shared__ unsigned short smem[4 * ABUF];   // [buf][A|B] : A(b)=b*2*ABUF, B(b)=+ABUF
  // Escape the LDS array: without this, alias analysis proves no instruction
  // ever stores to smem (TDM writes are invisible) and folds all reads to undef.
  { const unsigned short* esc = smem; LDS_ESCAPE(esc); }

  const int tid  = threadIdx.x;
  const int lane = tid & 31;
  const int wave = tid >> 5;
  const int m0 = blockIdx.y * 128;
  const int n0 = blockIdx.x * 128;
  const int wm = (wave & 3) * 32;
  const int wn = (wave >> 2) * 64;
  const int lm = lane & 15;
  const int half = lane >> 4;
  const unsigned stride_dw = (unsigned)(K >> 1);

  v8f acc[2][4];
  #pragma unroll
  for (int i = 0; i < 2; ++i)
    #pragma unroll
    for (int j = 0; j < 4; ++j)
      #pragma unroll
      for (int r = 0; r < 8; ++r) acc[i][j][r] = 0.f;

  const int nIter = K >> 5;
  if (wave == 0) {
    tdm_load_tile(0u,            A + (size_t)m0 * K, stride_dw);
    tdm_load_tile(2u * ABUF,     W + (size_t)n0 * K, stride_dw);
  }

  for (int it = 0; it < nIter; ++it) {
    const int cur = it & 1;
    if (wave == 0) {
      if (it + 1 < nIter) {
        const int kk2 = (it + 1) << 5;
        const unsigned nb = (unsigned)(cur ^ 1) * 4u * ABUF;   // byte off = elem*2
        tdm_load_tile(nb,            A + (size_t)m0 * K + kk2, stride_dw);
        tdm_load_tile(nb + 2u*ABUF,  W + (size_t)n0 * K + kk2, stride_dw);
        __builtin_amdgcn_s_wait_tensorcnt((short)2);   // current slab landed
      } else {
        __builtin_amdgcn_s_wait_tensorcnt((short)0);
      }
      LDS_CLOBBER();
    }
    __syncthreads();
    LDS_CLOBBER();

    const unsigned short* sAe = smem + cur * 2 * ABUF;
    const unsigned short* sBe = sAe + ABUF;

    FragB16 a[2], b[4];
    #pragma unroll
    for (int i = 0; i < 2; ++i) {
      int mrow = wm + i * 16 + lm;
      #pragma unroll
      for (int v = 0; v < 8; ++v) {
        int k = 2 * v + 8 * half + ((v >= 4) ? 8 : 0);   // A-frag k pairs
        a[i].u[v] = *(const unsigned int*)&sAe[mrow * 40 + k];
      }
    }
    #pragma unroll
    for (int j = 0; j < 4; ++j) {
      int nrow = wn + j * 16 + lm;
      #pragma unroll
      for (int v = 0; v < 8; ++v) {
        int k = 2 * v + 8 * half + ((v >= 4) ? 8 : 0);   // B-frag k pairs
        b[j].u[v] = *(const unsigned int*)&sBe[nrow * 40 + k];
      }
    }
    #pragma unroll
    for (int i = 0; i < 2; ++i)
      #pragma unroll
      for (int j = 0; j < 4; ++j)
        acc[i][j] = wmma_bf16(a[i], b[j], acc[i][j]);
    __syncthreads();
  }

  // Epilogue: C/D layout row = r + 8*half, col = lm
  #pragma unroll
  for (int i = 0; i < 2; ++i)
    #pragma unroll
    for (int j = 0; j < 4; ++j)
      #pragma unroll
      for (int r = 0; r < 8; ++r) {
        int row = m0 + wm + i * 16 + r + 8 * half;
        int col = n0 + wn + j * 16 + lm;
        size_t idx = (size_t)row * N + col;
        if (outF) outF[idx] = acc[i][j][r];
        else      outB[idx] = f2bf(acc[i][j][r]);
      }
}

// -------------------- RoPE + head split + V transpose --------------------
// qkv: [B*T, 3C] bf16. Outputs: qr,kr [B,H,T,hd] bf16 (rotated); vt [B,H,hd,T] bf16.
__global__ void k_rope(const unsigned short* __restrict__ qkv,
                       unsigned short* __restrict__ qr,
                       unsigned short* __restrict__ kr,
                       unsigned short* __restrict__ vt) {
  int idx = blockIdx.x * blockDim.x + threadIdx.x;   // B*H*T*64 threads
  int i = idx & 63;
  int t = (idx >> 6) & (TT - 1);
  int h = (idx >> 17) & (HH - 1);
  int b = idx >> 21;
  int bt = b * TT + t;
  int bh = b * HH + h;

  float inv = __powf(10000.f, -(float)(2 * i) / 128.f);
  float ang = (float)t * inv;
  float s, c;
  __sincosf(ang, &s, &c);

  const unsigned short* row = qkv + (size_t)bt * N_QKV;
  size_t base = ((size_t)bh * TT + t) * HD;
  // q
  {
    float x1 = bf2f(row[h * HD + 2 * i]);
    float x2 = bf2f(row[h * HD + 2 * i + 1]);
    qr[base + i]      = f2bf(x1 * c - x2 * s);
    qr[base + 64 + i] = f2bf(x1 * s + x2 * c);
  }
  // k
  {
    float x1 = bf2f(row[CC + h * HD + 2 * i]);
    float x2 = bf2f(row[CC + h * HD + 2 * i + 1]);
    kr[base + i]      = f2bf(x1 * c - x2 * s);
    kr[base + 64 + i] = f2bf(x1 * s + x2 * c);
  }
  // v transposed: [bh][d][t]
  {
    int d0 = 2 * i;
    vt[((size_t)bh * HD + d0)     * TT + t] = row[2 * CC + h * HD + d0];
    vt[((size_t)bh * HD + d0 + 1) * TT + t] = row[2 * CC + h * HD + d0 + 1];
  }
}

// -------------------- Flash attention (causal) --------------------
// Each wave: one 16-row q block. Grid: (T/128, B*H), 8 waves/block.
__global__ __launch_bounds__(256)
void k_attn(const unsigned short* __restrict__ qr,
            const unsigned short* __restrict__ kr,
            const unsigned short* __restrict__ vt,
            unsigned short* __restrict__ ao) {
  __shared__ unsigned short sP[8][16 * 40];
  const int lane = threadIdx.x & 31;
  const int wave = threadIdx.x >> 5;
  const int bh = blockIdx.y;
  const int b = bh >> 4, h = bh & 15;
  const int q0 = blockIdx.x * 128 + wave * 16;
  const int lm = lane & 15;
  const int half = lane >> 4;
  const float scale = 0.08838834764831845f;   // 1/sqrt(128)

  // Q fragments for all 4 K-chunks (hd = 128)
  FragB16 qa[4];
  const unsigned short* qrow = qr + ((size_t)bh * TT + q0 + lm) * HD;
  #pragma unroll
  for (int kc = 0; kc < 4; ++kc)
    #pragma unroll
    for (int v = 0; v < 8; ++v) {
      int k = kc * 32 + 2 * v + 8 * half + ((v >= 4) ? 8 : 0);
      qa[kc].u[v] = *(const unsigned int*)&qrow[k];
    }

  v8f o[8];
  #pragma unroll
  for (int dt = 0; dt < 8; ++dt)
    #pragma unroll
    for (int r = 0; r < 8; ++r) o[dt][r] = 0.f;
  float mrun[8], lrun[8];
  #pragma unroll
  for (int r = 0; r < 8; ++r) { mrun[r] = -1e30f; lrun[r] = 0.f; }

  for (int kv0 = 0; kv0 <= q0 + 15; kv0 += 32) {
    // prefetch next chunk of K and V into caches
    if (kv0 + 32 <= q0 + 15) {
      __builtin_prefetch(&kr[((size_t)bh * TT + kv0 + 32 + lm) * HD], 0, 1);
      __builtin_prefetch(&vt[((size_t)bh * HD + lane) * TT + kv0 + 32], 0, 1);
    }

    // ---- scores: two 16x16 tiles over 32 KV rows ----
    v8f s0, s1;
    #pragma unroll
    for (int r = 0; r < 8; ++r) { s0[r] = 0.f; s1[r] = 0.f; }
    #pragma unroll
    for (int kc = 0; kc < 4; ++kc) {
      FragB16 b0, b1;
      #pragma unroll
      for (int v = 0; v < 8; ++v) {
        int k = kc * 32 + 2 * v + 8 * half + ((v >= 4) ? 8 : 0);
        b0.u[v] = *(const unsigned int*)&kr[((size_t)bh * TT + kv0 + lm)      * HD + k];
        b1.u[v] = *(const unsigned int*)&kr[((size_t)bh * TT + kv0 + 16 + lm) * HD + k];
      }
      s0 = wmma_bf16(qa[kc], b0, s0);
      s1 = wmma_bf16(qa[kc], b1, s1);
    }

    // ---- mask + online softmax (rows r+8*half, col = lm within tile) ----
    #pragma unroll
    for (int r = 0; r < 8; ++r) {
      int rowg = q0 + r + 8 * half;
      float a0 = (kv0 + lm      <= rowg) ? s0[r] * scale : -1e30f;
      float a1 = (kv0 + 16 + lm <= rowg) ? s1[r] * scale : -1e30f;
      float mx = fmaxf(a0, a1);
      #pragma unroll
      for (int off = 1; off < 16; off <<= 1) mx = fmaxf(mx, __shfl_xor(mx, off, 32));
      float mnew  = fmaxf(mrun[r], mx);
      float alpha = __expf(mrun[r] - mnew);
      float p0 = __expf(a0 - mnew);
      float p1 = __expf(a1 - mnew);
      float sum = p0 + p1;
      #pragma unroll
      for (int off = 1; off < 16; off <<= 1) sum += __shfl_xor(sum, off, 32);
      lrun[r] = lrun[r] * alpha + sum;
      mrun[r] = mnew;
      #pragma unroll
      for (int dt = 0; dt < 8; ++dt) o[dt][r] *= alpha;
      // stage P tile (C-layout -> LDS, row-major 16x32)
      sP[wave][(r + 8 * half) * 40 + lm]      = f2bf(p0);
      sP[wave][(r + 8 * half) * 40 + 16 + lm] = f2bf(p1);
    }

    // reload P in A-fragment layout
    FragB16 pa;
    #pragma unroll
    for (int v = 0; v < 8; ++v) {
      int k = 2 * v + 8 * half + ((v >= 4) ? 8 : 0);
      pa.u[v] = *(const unsigned int*)&sP[wave][lm * 40 + k];
    }

    // ---- P @ V: 8 d-tiles of 16 ----
    #pragma unroll
    for (int dt = 0; dt < 8; ++dt) {
      FragB16 vb;
      #pragma unroll
      for (int v = 0; v < 8; ++v) {
        int k = kv0 + 2 * v + 8 * half + ((v >= 4) ? 8 : 0);     // t index, contiguous pair
        vb.u[v] = *(const unsigned int*)&vt[((size_t)bh * HD + dt * 16 + lm) * TT + k];
      }
      o[dt] = wmma_bf16(pa, vb, o[dt]);
    }
  }

  // ---- normalize + store to [B,T,C] bf16 ----
  #pragma unroll
  for (int r = 0; r < 8; ++r) {
    float inv = 1.f / lrun[r];
    int t = q0 + r + 8 * half;
    size_t base = ((size_t)b * TT + t) * CC + h * HD;
    #pragma unroll
    for (int dt = 0; dt < 8; ++dt)
      ao[base + dt * 16 + lm] = f2bf(o[dt][r] * inv);
  }
}

// -------------------- launch --------------------
extern "C" void kernel_launch(void* const* d_in, const int* in_sizes, int n_in,
                              void* d_out, int out_size, void* d_ws, size_t ws_size,
                              hipStream_t stream) {
  const float* x      = (const float*)d_in[0];
  const float* w_qkv  = (const float*)d_in[1];
  const float* w_proj = (const float*)d_in[2];
  // d_in[3]: causal mask (computed analytically in-kernel)
  float* out = (float*)d_out;

  char* ws = (char*)d_ws;
  size_t off = 0;
  unsigned short* xb   = (unsigned short*)(ws + off); off += (size_t)MM * CC * 2;      // 16 MB
  unsigned short* wqb  = (unsigned short*)(ws + off); off += (size_t)N_QKV * CC * 2;   // 24 MB
  unsigned short* wpb  = (unsigned short*)(ws + off); off += (size_t)CC * CC * 2;      //  8 MB
  unsigned short* qkvb = (unsigned short*)(ws + off); off += (size_t)MM * N_QKV * 2;   // 48 MB
  unsigned short* qrb  = (unsigned short*)(ws + off); off += (size_t)MM * CC * 2;      // 16 MB
  unsigned short* krb  = (unsigned short*)(ws + off); off += (size_t)MM * CC * 2;      // 16 MB
  unsigned short* vtb  = (unsigned short*)(ws + off); off += (size_t)MM * CC * 2;      // 16 MB
  unsigned short* aob  = (unsigned short*)(ws + off); off += (size_t)MM * CC * 2;      // 16 MB

  k_f32_to_bf16<<<2048, 256, 0, stream>>>(x,      xb,  MM * CC);
  k_f32_to_bf16<<<2048, 256, 0, stream>>>(w_qkv,  wqb, N_QKV * CC);
  k_f32_to_bf16<<<2048, 256, 0, stream>>>(w_proj, wpb, CC * CC);

  // qkv = x @ w_qkv^T  (bf16 out)
  k_gemm_bf16<<<dim3(N_QKV / 128, MM / 128), 256, 0, stream>>>(
      xb, wqb, nullptr, qkvb, MM, N_QKV, CC);

  // RoPE + head split + V transpose
  k_rope<<<(BB * HH * TT * 64) / 256, 256, 0, stream>>>(qkvb, qrb, krb, vtb);

  // causal flash attention
  k_attn<<<dim3(TT / 128, BB * HH), 256, 0, stream>>>(qrb, krb, vtb, aob);

  // final projection: out = attn @ w_proj^T  (fp32 out)
  k_gemm_bf16<<<dim3(CC / 128, MM / 128), 256, 0, stream>>>(
      aob, wpb, out, nullptr, MM, CC, CC);
}
